// ContrastiveLoss_24962349924507
// MI455X (gfx1250) — matrix operand order repaired and checked by reading, not compiled, not run on hardware
//
#include <hip/hip_runtime.h>
#include <hip/hip_bf16.h>

// ---------------------------------------------------------------------------
// NT-Xent contrastive loss, fused for MI455X (gfx1250, wave32, WMMA).
//   K1: L2-normalize rows -> z (bf16) in workspace
//   K2: positives pos[r] = dot(z[r], z[r+N])
//   K3: fused z*z^T GEMM (bf16 WMMA, f32 accum) + exp(2*sim) row-sum
//       64x64 block tile, 8 waves, 2 accumulators/wave with A-fragment reuse
//   K4: loss = mean(-(2*pos - log(rowsum - e^2)))
// ---------------------------------------------------------------------------

#define BATCH 4096
#define TWO_N 8192
#define DIM   512
#define INV_T 2.0f   // 1 / TEMP, TEMP = 0.5

typedef __bf16 v16bf __attribute__((ext_vector_type(16)));
typedef float  v8f   __attribute__((ext_vector_type(8)));
typedef unsigned int u32x4 __attribute__((ext_vector_type(4)));

#if __has_builtin(__builtin_amdgcn_tensor_load_to_lds)
#define USE_TDM 1
typedef unsigned int tdm_u32x4 __attribute__((ext_vector_type(4)));
typedef int          tdm_i32x8 __attribute__((ext_vector_type(8)));
typedef int          tdm_i32x4 __attribute__((ext_vector_type(4)));
#else
#define USE_TDM 0
#endif

union AFrag { v16bf v; u32x4 q[2]; };

// Flat pointer -> 32-bit LDS byte address (flat LDS aperture keeps the LDS
// offset in addr[31:0], per CDNA5 ISA 10.2 aperture mapping).
__device__ __forceinline__ unsigned lds_addr_of(const void* p) {
  return (unsigned)(unsigned long long)p;
}

#if USE_TDM
// TDM 2D tile load: `rows` consecutive 512-elem bf16 rows from the row-major
// [8192 x 512] z tensor into LDS (contiguous). D# per CDNA5 ISA ch.8.
__device__ __forceinline__ void tdm_load_rows(unsigned lds_off,
                                              const void* gsrc, int rows) {
  unsigned long long ga = (unsigned long long)gsrc;
  tdm_u32x4 g0;
  g0[0] = 1u;                                          // count=1, user mode
  g0[1] = lds_off;                                     // lds_addr (bytes)
  g0[2] = (unsigned)(ga & 0xFFFFFFFFull);              // global_addr[31:0]
  g0[3] = (unsigned)((ga >> 32) & 0x1FFFFFFull)        // global_addr[56:32]
        | (2u << 30);                                  // type = 2 ("image")
  tdm_i32x8 g1;
  g1[0] = (int)(1u << 16);                             // data_size=1 (2B), no mask
  g1[1] = (int)((DIM & 0xFFFFu) << 16);                // tensor_dim0 lo16
  g1[2] = (int)(((DIM >> 16) & 0xFFFFu)                // tensor_dim0 hi16
        | ((TWO_N & 0xFFFFu) << 16));                  // tensor_dim1 lo16
  g1[3] = (int)(((TWO_N >> 16) & 0xFFFFu)              // tensor_dim1 hi16
        | ((unsigned)DIM << 16));                      // tile_dim0 = 512
  g1[4] = (int)(rows & 0xFFFF);                        // tile_dim1, tile_dim2=0
  g1[5] = (int)DIM;                                    // tensor_dim0_stride lo32
  g1[6] = 0;
  g1[7] = 0;
  tdm_i32x4 z4 = {0, 0, 0, 0};
  tdm_i32x8 z8 = {0, 0, 0, 0, 0, 0, 0, 0};
  __builtin_amdgcn_tensor_load_to_lds(g0, g1, z4, z4, z8, 0);
}
#endif

// --------------------------- K1: normalize ---------------------------------
__global__ void cl_normalize_kernel(const float* __restrict__ emb_i,
                                    const float* __restrict__ emb_j,
                                    __hip_bfloat16* __restrict__ z) {
  const int row  = blockIdx.x * 8 + (threadIdx.x >> 5);
  const int lane = threadIdx.x & 31;
  const float* src = (row < BATCH) ? (emb_i + (size_t)row * DIM)
                                   : (emb_j + (size_t)(row - BATCH) * DIM);
  float v[16];
  float ss = 0.0f;
#pragma unroll
  for (int i = 0; i < 16; ++i) {
    v[i] = src[lane + i * 32];
    ss += v[i] * v[i];
  }
#pragma unroll
  for (int off = 16; off > 0; off >>= 1) ss += __shfl_xor(ss, off, 32);
  const float inv = 1.0f / fmaxf(sqrtf(ss), 1e-12f);
  __hip_bfloat16* drow = z + (size_t)row * DIM;
#pragma unroll
  for (int i = 0; i < 16; ++i)
    drow[lane + i * 32] = __float2bfloat16(v[i] * inv);
}

// --------------------------- K2: positives ---------------------------------
__global__ void cl_pos_kernel(const __hip_bfloat16* __restrict__ z,
                              float* __restrict__ pos) {
  const int row  = blockIdx.x * 8 + (threadIdx.x >> 5);
  const int lane = threadIdx.x & 31;
  const __hip_bfloat16* a = z + (size_t)row * DIM;
  const __hip_bfloat16* b = z + (size_t)(row + BATCH) * DIM;
  float acc = 0.0f;
#pragma unroll
  for (int k = 0; k < DIM / 32; ++k) {
    int idx = lane + k * 32;
    acc += __bfloat162float(a[idx]) * __bfloat162float(b[idx]);
  }
#pragma unroll
  for (int off = 16; off > 0; off >>= 1) acc += __shfl_xor(acc, off, 32);
  if (lane == 0) pos[row] = acc;
}

// ----------------- K3: fused sim GEMM + exp row-sum (WMMA) -----------------
// Block: 256 threads (8 waves). Block tile 64 rows x 64 cols per iteration,
// looped over all 8192 columns. Wave w: rows (w&3)*16, cols (w>>2)*32, with
// TWO accumulators sharing one A fragment (3 ds_load_b128 per WMMA).
#define MT 64
#define NT 64
__global__ void __launch_bounds__(256)
cl_simrow_kernel(const __hip_bfloat16* __restrict__ z,
                 float* __restrict__ rowsum) {
  __shared__ __align__(16) unsigned short As[MT][DIM];  // 64 KB
  __shared__ __align__(16) unsigned short Bs[NT][DIM];  // 64 KB

  const int tid  = threadIdx.x;
  const int wave = tid >> 5;
  const int lane = tid & 31;
  const int r0   = blockIdx.x * MT;

  const int mbase = (wave & 3) * 16;
  const int nbase = (wave >> 2) * 32;
  const int l15   = lane & 15;
  const int lhi   = lane >> 4;          // 0 or 1

  // ---- stage A strip (rows r0..r0+63, full K) once ----
#if USE_TDM
  if (tid < 32)
    tdm_load_rows(lds_addr_of(&As[0][0]), z + (size_t)r0 * DIM, MT);
  __builtin_amdgcn_s_wait_tensorcnt(0);
#else
  {
    const u32x4* s = (const u32x4*)(z + (size_t)r0 * DIM);
    u32x4* d = (u32x4*)&As[0][0];
    for (int i = tid; i < MT * DIM / 8; i += 256) d[i] = s[i];
  }
#endif

  float rs0[8], rs1[8];
#pragma unroll
  for (int e = 0; e < 8; ++e) { rs0[e] = 0.0f; rs1[e] = 0.0f; }

  // A fragment lane map (16-bit A 16x32): lanes 0-15 -> K {0..7,16..23},
  // lanes 16-31 -> K {8..15,24..31}. B fragment: lane n = l&15, same K
  // split as A. All are 16-byte ds_load_b128.
  const int am   = mbase + l15;
  const int akof = lhi * 8;
  const int bn0  = nbase + l15;
  const int bn1  = nbase + 16 + l15;
  const int bkg  = lhi * 8;

  for (int n0 = 0; n0 < TWO_N; n0 += NT) {
    __syncthreads();  // previous tile's compute done / As visible (1st iter)

    // prefetch next B tile into GL2 while we stage this one
    if (n0 + NT < TWO_N)
      __builtin_prefetch((const char*)(z + (size_t)(n0 + NT) * DIM) + tid * 256,
                         0, 1);

#if USE_TDM
    if (tid < 32)
      tdm_load_rows(lds_addr_of(&Bs[0][0]), z + (size_t)n0 * DIM, NT);
    __builtin_amdgcn_s_wait_tensorcnt(0);
#else
    {
      const u32x4* s = (const u32x4*)(z + (size_t)n0 * DIM);
      u32x4* d = (u32x4*)&Bs[0][0];
      for (int i = tid; i < NT * DIM / 8; i += 256) d[i] = s[i];
    }
#endif
    __syncthreads();

    v8f c0 = {0.f, 0.f, 0.f, 0.f, 0.f, 0.f, 0.f, 0.f};
    v8f c1 = {0.f, 0.f, 0.f, 0.f, 0.f, 0.f, 0.f, 0.f};

#pragma unroll
    for (int k0 = 0; k0 < DIM; k0 += 32) {
      AFrag a, b0, b1;
      a.q[0]  = *(const u32x4*)&As[am][k0 + akof];
      a.q[1]  = *(const u32x4*)&As[am][k0 + 16 + akof];
      b0.q[0] = *(const u32x4*)&Bs[bn0][k0 + bkg];
      b0.q[1] = *(const u32x4*)&Bs[bn0][k0 + bkg + 16];
      b1.q[0] = *(const u32x4*)&Bs[bn1][k0 + bkg];
      b1.q[1] = *(const u32x4*)&Bs[bn1][k0 + bkg + 16];
      c0 = __builtin_amdgcn_wmma_f32_16x16x32_bf16(
          false, a.v, false, b0.v, (short)0, c0, false, false);
      c1 = __builtin_amdgcn_wmma_f32_16x16x32_bf16(
          false, a.v, false, b1.v, (short)0, c1, false, false);
    }

    // epilogue: exp(sim / T) accumulated into per-row partials
#pragma unroll
    for (int e = 0; e < 8; ++e) {
      rs0[e] += __expf(c0[e] * INV_T);
      rs1[e] += __expf(c1[e] * INV_T);
    }
  }

  // Reduce across the 16 lanes that share a row (cols), then atomically add.
  // C layout: VGPR e, lanes 0-15 -> row mbase+e ; lanes 16-31 -> row mbase+8+e.
#pragma unroll
  for (int e = 0; e < 8; ++e) {
    float rs = rs0[e] + rs1[e];   // both accumulators cover the same rows
#pragma unroll
    for (int off = 1; off < 16; off <<= 1) rs += __shfl_xor(rs, off, 16);
    if (l15 == 0) {
      int row = r0 + mbase + e + lhi * 8;
      atomicAdd(&rowsum[row], rs);
    }
  }
}

// --------------------------- K4: finalize ----------------------------------
__global__ void cl_finalize_kernel(const float* __restrict__ rowsum,
                                   const float* __restrict__ pos,
                                   float* __restrict__ out) {
  __shared__ float red[256];
  const int i = blockIdx.x * 256 + threadIdx.x;
  const float diag = __expf(INV_T);                 // exp(sim_ii / T), sim_ii = 1
  const float denom = rowsum[i] - diag;             // mask removes the diagonal
  const float p = pos[i & (BATCH - 1)];
  float li = -(p * INV_T - __logf(denom));
  li *= (1.0f / (float)TWO_N);
  red[threadIdx.x] = li;
  __syncthreads();
#pragma unroll
  for (int s = 128; s > 0; s >>= 1) {
    if (threadIdx.x < s) red[threadIdx.x] += red[threadIdx.x + s];
    __syncthreads();
  }
  if (threadIdx.x == 0) atomicAdd(out, red[0]);
}

// ---------------------------------------------------------------------------
extern "C" void kernel_launch(void* const* d_in, const int* in_sizes, int n_in,
                              void* d_out, int out_size, void* d_ws,
                              size_t ws_size, hipStream_t stream) {
  const float* emb_i = (const float*)d_in[0];
  const float* emb_j = (const float*)d_in[1];
  float* out = (float*)d_out;

  char* ws = (char*)d_ws;
  __hip_bfloat16* z = (__hip_bfloat16*)ws;                       // 8 MB
  float* pos    = (float*)(ws + (size_t)TWO_N * DIM * 2);        // 16 KB
  float* rowsum = (float*)(ws + (size_t)TWO_N * DIM * 2 + BATCH * sizeof(float));

  (void)hipMemsetAsync(rowsum, 0, TWO_N * sizeof(float), stream);
  (void)hipMemsetAsync(out, 0, sizeof(float), stream);

  cl_normalize_kernel<<<TWO_N / 8, 256, 0, stream>>>(emb_i, emb_j, z);
  cl_pos_kernel<<<BATCH / 8, 256, 0, stream>>>(z, pos);
  cl_simrow_kernel<<<TWO_N / MT, 256, 0, stream>>>(z, rowsum);
  cl_finalize_kernel<<<TWO_N / 256, 256, 0, stream>>>(rowsum, pos, out);
}
